// LSTM_27943057227794
// MI455X (gfx1250) — matrix operand order repaired and checked by reading, not compiled, not run on hardware
//
#include <hip/hip_runtime.h>
#include <hip/hip_fp16.h>

typedef __attribute__((ext_vector_type(16))) _Float16 v16h;
typedef __attribute__((ext_vector_type(8)))  _Float16 v8h;
typedef __attribute__((ext_vector_type(4)))  _Float16 v4h;
typedef __attribute__((ext_vector_type(8)))  float    v8f;
typedef __attribute__((ext_vector_type(4)))  float    v4f;

#define T_STEPS 2048
#define BATCH   64
#define INDIM   256
#define HID     512
#define KDIM    768       // HID + INDIM (cat order: [h, x])
#define NWGX    16        // column-slice workgroups per batch group
#define NWGY    4         // batch groups (16 rows each)
#define BG      16        // batch rows per group  (== WMMA M)
#define COLS    128       // gate-columns per WG (4 gates * 32 units)
#define UNITS   32        // hidden units per WG
#define WSTR    776       // padded LDS row stride in halves (768 + 8)
#define NTHREADS 256      // 8 waves (wave32)

// LDS budget (bytes): W slice 128*776*2 = 198656, A 16*776*2 = 24832,
// bias 128*4 = 512, gate xchg 4*16*32*4 = 8192, c-state 16*32*4 = 2048
#define SMEM_BYTES (((COLS*WSTR + BG*WSTR) * 2) + ((COLS + 4*BG*UNITS + BG*UNITS) * 4))

// Fast activations: single v_exp_f32 + v_rcp_f32, no IEEE-div expansion.
__device__ __forceinline__ float fast_sigmoid(float v) {
  return __builtin_amdgcn_rcpf(1.0f + __expf(-v));
}
__device__ __forceinline__ float fast_tanh(float v) {
  // tanh(v) = 1 - 2/(exp(2v)+1); exp->0 gives -1, exp->inf gives +1.
  return 1.0f - 2.0f * __builtin_amdgcn_rcpf(1.0f + __expf(2.0f * v));
}

__global__ __launch_bounds__(NTHREADS, 1)
void lstm_persistent(const float* __restrict__ x,
                     const float* __restrict__ Wf, const float* __restrict__ bf,
                     const float* __restrict__ Wi, const float* __restrict__ bi,
                     const float* __restrict__ Wc, const float* __restrict__ bc,
                     const float* __restrict__ Wo, const float* __restrict__ bo,
                     float* __restrict__ out,            // [T][B][H] ++ h_last ++ c_last
                     unsigned int* __restrict__ counters,// [NWGY]
                     _Float16* __restrict__ hbuf)        // [2][B][H] f16 ping-pong
{
  extern __shared__ char smem_raw[];
  _Float16* Wl = (_Float16*)smem_raw;          // [COLS][WSTR]
  _Float16* Al = Wl + COLS * WSTR;             // [BG][WSTR]   (hx staging)
  float*    Bl = (float*)(Al + BG * WSTR);     // [COLS]       (bias)
  float*    Gl = Bl + COLS;                    // [4][BG][UNITS] gate exchange
  float*    Cl = Gl + 4 * BG * UNITS;          // [BG][UNITS]  c-state

  const int tid = threadIdx.x;
  const int wx  = blockIdx.x;                  // column slice 0..15
  const int wy  = blockIdx.y;                  // batch group 0..3
  const int b0  = wy * BG;
  const int h0  = wx * UNITS;

  const float* Wg[4]  = {Wf, Wi, Wc, Wo};
  const float* bgv[4] = {bf, bi, bc, bo};

  // ---- one-time: convert weight slice to f16 in LDS (reused for 2048 steps) ----
  for (int r = 0; r < COLS; ++r) {
    const int g = r >> 5, u = r & 31;
    const float* src = Wg[g] + (size_t)(h0 + u) * KDIM;
    for (int k = tid; k < KDIM; k += NTHREADS)
      Wl[r * WSTR + k] = (_Float16)src[k];
  }
  if (tid < COLS) Bl[tid] = bgv[tid >> 5][h0 + (tid & 31)];
  for (int i2 = tid; i2 < BG * UNITS; i2 += NTHREADS) Cl[i2] = 0.0f;
  __syncthreads();

  const int lane  = tid & 31;
  const int wv    = tid >> 5;      // wave id == output n-tile (0..7)
  const int nlo   = lane & 15;     // N within tile (also A row M for frag loads)
  const int hiGrp = lane >> 4;     // lane-half selector for K packing

  // Gate index is wave-uniform (each wave's 16 columns sit inside one 32-col
  // gate block). Force it into an SGPR so activation select is a scalar branch.
  const int gwave = __builtin_amdgcn_readfirstlane(wv >> 1);   // 0=f 1=i 2=c 3=o
  const int uwave = (wv & 1) * 16 + nlo;                       // unit within slice

  // Physical cluster id: 0 => not dispatched in a cluster (use atomic barrier)
  unsigned int clus;
  asm volatile("s_getreg_b32 %0, hwreg(HW_REG_IB_STS2, 6, 4)" : "=s"(clus));

  unsigned int* cnt = counters + wy;

  for (int t = 0; t < T_STEPS; ++t) {
    // ---- stage A = [h_prev (f16) | x_t (f32->f16)] into LDS ----
    const _Float16* hsrc = hbuf + (size_t)(t & 1) * BATCH * HID;
    for (int idx = tid; idx < BG * (HID / 8); idx += NTHREADS) {
      const int m = idx >> 6, c8 = idx & 63;
      *(v8h*)(Al + m * WSTR + c8 * 8) = *(const v8h*)(hsrc + (b0 + m) * HID + c8 * 8);
    }
    const float* xsrc = x + (size_t)t * BATCH * INDIM;
    for (int idx = tid; idx < BG * (INDIM / 4); idx += NTHREADS) {
      const int m = idx >> 6, c4 = idx & 63;
      v4f xv = *(const v4f*)(xsrc + (b0 + m) * INDIM + c4 * 4);
      v4h hv;
      hv.x = (_Float16)xv.x; hv.y = (_Float16)xv.y;
      hv.z = (_Float16)xv.z; hv.w = (_Float16)xv.w;
      *(v4h*)(Al + m * WSTR + HID + c4 * 4) = hv;
    }
    if (t + 1 < T_STEPS && tid < BG)   // pull next x slice toward L2
      __builtin_prefetch(xsrc + BATCH * INDIM + (b0 + tid) * INDIM, 0, 1);
    __syncthreads();

    // ---- GEMM: wave wv -> 16x16 tile of columns [wv*16, wv*16+16), K = 768 ----
    const float bsc = Bl[wv * 16 + nlo];       // bias broadcast into all 8 C rows
    v8f acc;
#pragma unroll
    for (int r = 0; r < 8; ++r) acc[r] = bsc;

    const _Float16* aptr = Al + nlo * WSTR + hiGrp * 8;
    const _Float16* bptr = Wl + (wv * 16 + nlo) * WSTR + hiGrp * 8;
#pragma unroll 4
    for (int kt = 0; kt < KDIM / 32; ++kt) {
      v8h a0  = *(const v8h*)(aptr);           // K {0..7}  (+8 for upper lanes)
      v8h a1  = *(const v8h*)(aptr + 16);      // K {16..23}(+8 for upper lanes)
      v8h bb0 = *(const v8h*)(bptr);           // K {0..7}  (+8 for upper lanes)
      v8h bb1 = *(const v8h*)(bptr + 16);      // K {16..23}(+8 for upper lanes)
      v16h av = __builtin_shufflevector(a0, a1, 0,1,2,3,4,5,6,7,8,9,10,11,12,13,14,15);
      v16h bv = __builtin_shufflevector(bb0, bb1, 0,1,2,3,4,5,6,7,8,9,10,11,12,13,14,15);
      acc = __builtin_amdgcn_wmma_f32_16x16x32_f16(false, av, false, bv,
                                                   (short)0, acc, false, false);
      aptr += 32;
      bptr += 32;
    }

    // ---- activation (scalar-branched per wave), scatter into gate-exchange LDS ----
    {
      float* gslot = Gl + gwave * (BG * UNITS) + uwave;
      if (gwave == 2) {
#pragma unroll
        for (int r = 0; r < 8; ++r)
          gslot[(r + hiGrp * 8) * UNITS] = fast_tanh(acc[r]);
      } else {
#pragma unroll
        for (int r = 0; r < 8; ++r)
          gslot[(r + hiGrp * 8) * UNITS] = fast_sigmoid(acc[r]);
      }
    }
    __syncthreads();

    // ---- elementwise c/h update; emit output (NT stores) + next-h (f16) ----
    _Float16* hdst = hbuf + (size_t)((t + 1) & 1) * BATCH * HID;
    for (int idx = tid; idx < BG * UNITS; idx += NTHREADS) {
      const int m = idx >> 5, u = idx & 31;
      const float fg = Gl[0 * BG * UNITS + idx];
      const float ig = Gl[1 * BG * UNITS + idx];
      const float cb = Gl[2 * BG * UNITS + idx];
      const float og = Gl[3 * BG * UNITS + idx];
      const float cn = fg * Cl[idx] + ig * cb;
      const float hn = og * fast_tanh(cn);
      Cl[idx] = cn;
      // 268 MB output stream > 192 MB L2: keep it out of the cache (TH=NT).
      __builtin_nontemporal_store(hn, &out[((size_t)t * BATCH + (b0 + m)) * HID + (h0 + u)]);
      hdst[(b0 + m) * HID + (h0 + u)] = (_Float16)hn;
      if (t == T_STEPS - 1) {
        const size_t base = (size_t)T_STEPS * BATCH * HID;
        out[base + (size_t)(b0 + m) * HID + (h0 + u)] = hn;                       // h_last
        out[base + (size_t)BATCH * HID + (size_t)(b0 + m) * HID + (h0 + u)] = cn; // c_last
      }
    }

    // ---- inter-WG barrier for this batch group's recurrence ----
    __threadfence();          // release h stores to device scope
    __syncthreads();
    if (clus != 0) {
      // CDNA5 cluster split-barrier: one wave signals, all waves wait
      if (wv == 0) asm volatile("s_barrier_signal -3" ::: "memory");
      asm volatile("s_barrier_wait -3" ::: "memory");
    } else {
      if (tid == 0) {
        __hip_atomic_fetch_add(cnt, 1u, __ATOMIC_RELEASE, __HIP_MEMORY_SCOPE_AGENT);
        const unsigned int tgt = (unsigned int)(NWGX * (t + 1));
        while (__hip_atomic_load(cnt, __ATOMIC_ACQUIRE, __HIP_MEMORY_SCOPE_AGENT) < tgt)
          __builtin_amdgcn_s_sleep(1);
      }
    }
    __builtin_amdgcn_fence(__ATOMIC_ACQUIRE, "agent"); // invalidate stale h lines
    __syncthreads();
  }
}

__global__ void lstm_init(unsigned int* counters, _Float16* hbuf) {
  const int idx = blockIdx.x * blockDim.x + threadIdx.x;
  if (idx < NWGY) counters[idx] = 0u;
  const int total = 2 * BATCH * HID / 8;       // zero both h ping-pong buffers
  v8h z = {};
  for (int i = idx; i < total; i += (int)(gridDim.x * blockDim.x))
    *(v8h*)(hbuf + i * 8) = z;
}

extern "C" void kernel_launch(void* const* d_in, const int* in_sizes, int n_in,
                              void* d_out, int out_size, void* d_ws, size_t ws_size,
                              hipStream_t stream) {
  (void)in_sizes; (void)n_in; (void)out_size; (void)ws_size;
  const float* x  = (const float*)d_in[0];
  const float* Wf = (const float*)d_in[1];
  const float* bf = (const float*)d_in[2];
  const float* Wi = (const float*)d_in[3];
  const float* bi = (const float*)d_in[4];
  const float* Wc = (const float*)d_in[5];
  const float* bc = (const float*)d_in[6];
  const float* Wo = (const float*)d_in[7];
  const float* bo = (const float*)d_in[8];

  unsigned int* counters = (unsigned int*)d_ws;
  _Float16* hbuf = (_Float16*)((char*)d_ws + 256);

  lstm_init<<<dim3(32), dim3(256), 0, stream>>>(counters, hbuf);

  // 234,240 B of dynamic LDS per WG (< 320 KB WGP LDS); opt in above 64 KB.
  hipFuncSetAttribute(reinterpret_cast<const void*>(&lstm_persistent),
                      hipFuncAttributeMaxDynamicSharedMemorySize, (int)SMEM_BYTES);

  lstm_persistent<<<dim3(NWGX, NWGY), dim3(NTHREADS), SMEM_BYTES, stream>>>(
      x, Wf, bf, Wi, bi, Wc, bc, Wo, bo,
      (float*)d_out, counters, hbuf);
}
